// BahdanauAttention_73091753443736
// MI455X (gfx1250) — matrix operand order repaired and checked by reading, not compiled, not run on hardware
//
#include <hip/hip_runtime.h>
#include <hip/hip_bf16.h>

typedef __attribute__((ext_vector_type(16))) __bf16 v16bf;
typedef __attribute__((ext_vector_type(8)))  float  v8f;

#define B_DIM 64
#define T_DIM 2048
#define D_DIM 256
#define U_DIM 256
#define W1_PAD 8
#define W1_ROW (D_DIM + W1_PAD)   // 264 bf16 per row -> 528B stride, 16B aligned, bank-spread

union ABFrag {
    v16bf  v;
    __bf16 e[16];
};

__device__ __forceinline__ void cvt8(__bf16* dst, float4 a, float4 b) {
    dst[0] = (__bf16)a.x; dst[1] = (__bf16)a.y; dst[2] = (__bf16)a.z; dst[3] = (__bf16)a.w;
    dst[4] = (__bf16)b.x; dst[5] = (__bf16)b.y; dst[6] = (__bf16)b.z; dst[7] = (__bf16)b.w;
}

__device__ __forceinline__ float fast_tanhf(float x) {
    // tanh(x) = 1 - 2/(exp(2x)+1) ; v_exp_f32 + v_rcp_f32
    float e = __expf(2.0f * x);
    return 1.0f - 2.0f * __builtin_amdgcn_rcpf(e + 1.0f);
}

// ---------------------------------------------------------------------------
// Kernel 1: hb[b,u] = hidden[b,:]·w2_w[u,:] + w2_b[u] + w1_b[u]
// ---------------------------------------------------------------------------
__global__ __launch_bounds__(256) void hid_kernel(
    const float* __restrict__ hidden, const float* __restrict__ w2_w,
    const float* __restrict__ w2_b, const float* __restrict__ w1_b,
    float* __restrict__ hb)
{
    int b = blockIdx.x, u = threadIdx.x;
    const float4* h = (const float4*)(hidden + (size_t)b * D_DIM);
    const float4* w = (const float4*)(w2_w + (size_t)u * D_DIM);
    float acc = 0.0f;
#pragma unroll 8
    for (int i = 0; i < D_DIM / 4; ++i) {
        float4 x = h[i], y = w[i];
        acc += x.x * y.x + x.y * y.y + x.z * y.z + x.w * y.w;
    }
    hb[(size_t)b * U_DIM + u] = acc + w2_b[u] + w1_b[u];
}

// ---------------------------------------------------------------------------
// Kernel 2: scores[b,t] = v_w · tanh(features·w1ᵀ + hb[b]) + v_b   (WMMA bf16)
// 256 blocks x 256 threads (8 waves); each wave: one 16-row M tile per iter.
// ---------------------------------------------------------------------------
__global__ __launch_bounds__(256) void score_kernel(
    const float* __restrict__ feat,   // [B*T, 256]
    const float* __restrict__ w1_w,   // [256, 256]
    const float* __restrict__ hb,     // [B, 256]
    const float* __restrict__ v_w,    // [256]
    const float* __restrict__ v_b,    // [1]
    float* __restrict__ scores)       // [B*T]
{
    extern __shared__ __align__(16) char smem[];
    __bf16* w1s = (__bf16*)smem;      // [256][W1_ROW] bf16

    int tid = threadIdx.x;

    // Cooperative load of w1 into LDS as bf16 (coalesced float4 reads).
    for (int i = tid; i < U_DIM * (D_DIM / 4); i += 256) {
        int u  = i >> 6;
        int d4 = (i & 63) << 2;
        float4 f = ((const float4*)w1_w)[i];
        __bf16* dst = &w1s[(size_t)u * W1_ROW + d4];
        dst[0] = (__bf16)f.x; dst[1] = (__bf16)f.y;
        dst[2] = (__bf16)f.z; dst[3] = (__bf16)f.w;
    }
    __syncthreads();

    int lane = tid & 31;
    int wv   = tid >> 5;                 // wave 0..7
    int r    = lane & 15;                // A row within tile
    int kh   = (lane >> 4) << 3;         // A half-K offset: 0 or 8
    int bk   = (lane >> 4) << 3;         // B half-K offset: 0 or 8
    int c    = lane & 15;                // B/C column
    float vb = v_b[0];

#pragma unroll 1
    for (int it = 0; it < 4; ++it) {
        int tile = ((it * (int)gridDim.x) + (int)blockIdx.x) * 8 + wv;  // 0..8191
        size_t R = (size_t)tile * 16;                                   // flat row base
        int b = (int)(R >> 11);                                         // T = 2048

        // Load the full 16x256 A tile (fp32 -> bf16), 8 K-step fragments.
        ABFrag a[8];
        const float* rowp = feat + (R + (size_t)r) * D_DIM;
#pragma unroll
        for (int ks = 0; ks < 8; ++ks) {
            const float* p = rowp + ks * 32 + kh;
            float4 f0 = ((const float4*)p)[0];
            float4 f1 = ((const float4*)p)[1];
            float4 f2 = ((const float4*)(p + 16))[0];
            float4 f3 = ((const float4*)(p + 16))[1];
            cvt8(&a[ks].e[0], f0, f1);   // K = kh+0..7
            cvt8(&a[ks].e[8], f2, f3);   // K = kh+16..23
        }

        float sAcc[8] = {0, 0, 0, 0, 0, 0, 0, 0};
        const float* hbrow = hb + (size_t)b * U_DIM;

#pragma unroll 1
        for (int n = 0; n < 16; ++n) {
            v8f acc = {};
            int u = (n << 4) + c;
            const __bf16* wrow = &w1s[(size_t)u * W1_ROW + bk];
#pragma unroll
            for (int ks = 0; ks < 8; ++ks) {
                ABFrag bb;
                const __bf16* bp = wrow + ks * 32;
                *(float4*)&bb.e[0] = *(const float4*)bp;        // K = bk+0..7
                *(float4*)&bb.e[8] = *(const float4*)(bp + 16); // K = bk+16..23
                acc = __builtin_amdgcn_wmma_f32_16x16x32_bf16(
                    false, a[ks].v, false, bb.v, (short)0, acc, false, false);
            }
            float su = hbrow[u];
            float vw = v_w[u];
#pragma unroll
            for (int v = 0; v < 8; ++v) {
                sAcc[v] += fast_tanhf(acc[v] + su) * vw;
            }
        }

        // Reduce over the 16 lanes of each half-wave -> per-row score.
#pragma unroll
        for (int v = 0; v < 8; ++v) {
            float s = sAcc[v];
            s += __shfl_xor(s, 1, 32);
            s += __shfl_xor(s, 2, 32);
            s += __shfl_xor(s, 4, 32);
            s += __shfl_xor(s, 8, 32);
            if (c == 0) {
                int row = (lane >> 4) ? (v + 8) : v;
                scores[R + row] = s + vb;
            }
        }
    }
}

// ---------------------------------------------------------------------------
// Kernel 3: softmax over T, in place on scores region of d_out.
// ---------------------------------------------------------------------------
__global__ __launch_bounds__(256) void softmax_kernel(float* __restrict__ sc)
{
    __shared__ float red[8];
    int b = blockIdx.x, tid = threadIdx.x;
    float* s = sc + (size_t)b * T_DIM;

    float mx = -3.4e38f;
    for (int t = tid; t < T_DIM; t += 256) mx = fmaxf(mx, s[t]);
    mx = fmaxf(mx, __shfl_xor(mx, 1, 32));
    mx = fmaxf(mx, __shfl_xor(mx, 2, 32));
    mx = fmaxf(mx, __shfl_xor(mx, 4, 32));
    mx = fmaxf(mx, __shfl_xor(mx, 8, 32));
    mx = fmaxf(mx, __shfl_xor(mx, 16, 32));
    if ((tid & 31) == 0) red[tid >> 5] = mx;
    __syncthreads();
    float m = red[0];
#pragma unroll
    for (int i = 1; i < 8; ++i) m = fmaxf(m, red[i]);
    __syncthreads();

    float sum = 0.0f;
    for (int t = tid; t < T_DIM; t += 256) {
        float e = __expf(s[t] - m);
        s[t] = e;
        sum += e;
    }
    sum += __shfl_xor(sum, 1, 32);
    sum += __shfl_xor(sum, 2, 32);
    sum += __shfl_xor(sum, 4, 32);
    sum += __shfl_xor(sum, 8, 32);
    sum += __shfl_xor(sum, 16, 32);
    if ((tid & 31) == 0) red[tid >> 5] = sum;
    __syncthreads();
    float tot = 0.0f;
#pragma unroll
    for (int i = 0; i < 8; ++i) tot += red[i];
    float inv = 1.0f / tot;
    for (int t = tid; t < T_DIM; t += 256) s[t] *= inv;
}

// ---------------------------------------------------------------------------
// Kernel 4: partial context sums  part[b,chunk,d] = sum_{t in chunk} attn*feat
// ---------------------------------------------------------------------------
__global__ __launch_bounds__(256) void ctx_partial(
    const float* __restrict__ feat, const float* __restrict__ attn,
    float* __restrict__ part)
{
    __shared__ float aw[256];
    int b = blockIdx.x, ch = blockIdx.y, d = threadIdx.x;
    int t0 = ch * 256;
    aw[d] = attn[(size_t)b * T_DIM + t0 + d];
    __syncthreads();
    const float* fp = feat + ((size_t)b * T_DIM + t0) * D_DIM + d;
    float acc = 0.0f;
#pragma unroll 4
    for (int t = 0; t < 256; ++t) acc += aw[t] * fp[(size_t)t * D_DIM];
    part[((size_t)b * 8 + ch) * D_DIM + d] = acc;
}

// ---------------------------------------------------------------------------
// Kernel 5: reduce partials -> context (deterministic, no fp atomics)
// ---------------------------------------------------------------------------
__global__ __launch_bounds__(256) void ctx_reduce(
    const float* __restrict__ part, float* __restrict__ ctx)
{
    int b = blockIdx.x, d = threadIdx.x;
    float acc = 0.0f;
#pragma unroll
    for (int c = 0; c < 8; ++c) acc += part[((size_t)b * 8 + c) * D_DIM + d];
    ctx[(size_t)b * D_DIM + d] = acc;
}

// ---------------------------------------------------------------------------
extern "C" void kernel_launch(void* const* d_in, const int* in_sizes, int n_in,
                              void* d_out, int out_size, void* d_ws, size_t ws_size,
                              hipStream_t stream) {
    (void)in_sizes; (void)n_in; (void)out_size; (void)ws_size;
    const float* features = (const float*)d_in[0];   // [B,T,D]
    const float* hidden   = (const float*)d_in[1];   // [B,D]
    const float* w1_w     = (const float*)d_in[2];   // [U,D]
    const float* w1_b     = (const float*)d_in[3];   // [U]
    const float* w2_w     = (const float*)d_in[4];   // [U,D]
    const float* w2_b     = (const float*)d_in[5];   // [U]
    const float* v_w      = (const float*)d_in[6];   // [1,U]
    const float* v_b      = (const float*)d_in[7];   // [1]

    float* out    = (float*)d_out;
    float* ctx    = out;                       // [B,D]    = 16384 floats
    float* scores = out + B_DIM * D_DIM;       // [B,T]    = 131072 floats (attn)

    float* hb   = (float*)d_ws;                // [B,U]   64KB
    float* part = hb + B_DIM * U_DIM;          // [B,8,D] 512KB

    hid_kernel<<<B_DIM, 256, 0, stream>>>(hidden, w2_w, w2_b, w1_b, hb);

    size_t w1_lds = (size_t)U_DIM * W1_ROW * sizeof(__bf16);  // 135168 B
    score_kernel<<<256, 256, w1_lds, stream>>>(features, w1_w, hb, v_w, v_b, scores);

    softmax_kernel<<<B_DIM, 256, 0, stream>>>(scores);

    ctx_partial<<<dim3(B_DIM, 8), 256, 0, stream>>>(features, scores, part);

    ctx_reduce<<<B_DIM, 256, 0, stream>>>(part, ctx);
}
